// PointEncoder_28879360098443
// MI455X (gfx1250) — matrix operand order repaired and checked by reading, not compiled, not run on hardware
//
#include <hip/hip_runtime.h>

typedef __attribute__((ext_vector_type(16))) _Float16 v16h;
typedef __attribute__((ext_vector_type(8)))  float    v8f;

#define HID 64
#define NLYR 4
#define NPTS 4096
#define NBATCH 256
#define WAVES 8
#define THREADS 256
#define SLABS 2
#define PTS_PER_WG (WAVES * 16 * SLABS)   // 256
#define CHUNKS (NPTS / PTS_PER_WG)        // 16

union U16x8 { v16h v; unsigned u[8]; };

__device__ __forceinline__ unsigned pkh(float a, float b) {
  union { _Float16 h[2]; unsigned u; } t;
  t.h[0] = (_Float16)a; t.h[1] = (_Float16)b; return t.u;
}

// A fragments (16x32 f16, ISA 7.12.2) from row-major [16][64] f16 LDS slab.
__device__ __forceinline__ void loadA(const _Float16* lds, int lane, v16h& A0, v16h& A1) {
  const unsigned* lds32 = (const unsigned*)lds;
  int m  = lane & 15;
  int hi = (lane >> 4) * 8;
  U16x8 a0, a1;
#pragma unroll
  for (int v = 0; v < 8; ++v) {
    int koff = ((v >> 2) * 16) + hi + ((v & 3) * 2);
    a0.u[v] = lds32[(m * 64 + koff) >> 1];        // K 0..31
    a1.u[v] = lds32[(m * 64 + 32 + koff) >> 1];   // K 32..63
  }
  A0 = a0.v; A1 = a1.v;
}

// Stage one 64x64 weight block (B[k,n] = W[nt*16+n][kofs+k]) into fragment-order
// packed-f16 LDS: dst[frag][lane][v], frag = nt*2 + (kofs==32).
__device__ __forceinline__ void stageB(unsigned* dst, const float* __restrict__ W,
                                       int ldw, int tid) {
  for (int widx = tid; widx < 8 * 32 * 8; widx += THREADS) {
    int frag = widx >> 8;
    int lane = (widx >> 3) & 31;
    int v    = widx & 7;
    int nt   = frag >> 1;
    int kofs = (frag & 1) * 32;
    int n    = lane & 15;
    int k    = ((v >> 2) * 16) + ((lane >> 4) * 8) + kofs + 2 * (v & 3);
    const float* p = W + (size_t)(nt * 16 + n) * ldw + k;
    dst[widx] = pkh(p[0], p[1]);
  }
}

// Fetch a staged B fragment: two ds_load_b128, zero VALU.
__device__ __forceinline__ v16h loadBst(const unsigned* base, int frag, int lane) {
  const uint4* p = (const uint4*)(base + ((frag * 32 + lane) << 3));
  uint4 lo = p[0], hi = p[1];
  U16x8 b;
  b.u[0] = lo.x; b.u[1] = lo.y; b.u[2] = lo.z; b.u[3] = lo.w;
  b.u[4] = hi.x; b.u[5] = hi.y; b.u[6] = hi.z; b.u[7] = hi.w;
  return b.v;
}

// C tile (16x16 f32) -> relu -> f16 row-major LDS slab
__device__ __forceinline__ void storeC(_Float16* lds, int lane, int col, v8f c) {
  int n  = lane & 15;
  int mb = (lane >> 4) * 8;
#pragma unroll
  for (int r = 0; r < 8; ++r) {
    float f = c[r]; f = f > 0.f ? f : 0.f;
    lds[(mb + r) * 64 + col + n] = (_Float16)f;
  }
}

__device__ __forceinline__ v8f wmma(v16h a, v16h b, v8f c) {
  return __builtin_amdgcn_wmma_f32_16x16x32_f16(false, a, false, b, (short)0, c, false, false);
}

__global__ void __launch_bounds__(THREADS)
pe_pass(const float* __restrict__ x, const float* __restrict__ w_in, const float* __restrict__ b_in,
        const float* __restrict__ lw, const float* __restrict__ lb,
        const float* __restrict__ gw, const float* __restrict__ gbv,
        const float* __restrict__ wo,
        unsigned* __restrict__ g_ws, unsigned* __restrict__ o_ws, int pass)
{
  __shared__ __align__(16) _Float16 ldsH[WAVES * SLABS * 16 * 64]; // 32 KB relayout scratch
  __shared__ __align__(16) unsigned Bst[3][8][32][8];              // 24 KB staged B frags
  __shared__ float    gbias[NLYR][HID];
  __shared__ unsigned red[HID];

  const int tid   = threadIdx.x;
  const int lane  = tid & 31;
  const int wave  = tid >> 5;
  const int b     = blockIdx.y;
  const int chunk = blockIdx.x;

  // gbias[i][o] = glyr_b[i][o] + sum_c g_i[c] * Wg[i][o][64+c]  (g_i known for i < pass)
  const int ngl = pass < NLYR ? pass : NLYR;
  for (int idx = tid; idx < ngl * HID; idx += THREADS) {
    int i = idx >> 6, o = idx & 63;
    float s = gbv[i * HID + o];
    const float*    wr = gw + (size_t)i * HID * 2 * HID + (size_t)o * 2 * HID + HID;
    const unsigned* gi = g_ws + ((size_t)i * NBATCH + b) * HID;
    for (int c = 0; c < HID; ++c) s += __uint_as_float(gi[c]) * wr[c];
    gbias[i][o] = s;
  }
  if (tid < HID) red[tid] = 0u;

  _Float16* myH[SLABS];
#pragma unroll
  for (int s = 0; s < SLABS; ++s) myH[s] = ldsH + (wave * SLABS + s) * (16 * 64);
  const int p0 = chunk * PTS_PER_WG + wave * (16 * SLABS);

  // ---- proj_in + relu into LDS slabs (K=3: VALU, negligible FLOPs) ----
#pragma unroll
  for (int s = 0; s < SLABS; ++s) {
    const float* xb = x + ((size_t)b * NPTS + p0 + s * 16) * 3;
    for (int idx = lane; idx < 16 * 64; idx += 32) {
      int m = idx >> 6, k = idx & 63;
      const float* xm = xb + m * 3;
      float v = b_in[k] + xm[0] * w_in[k * 3] + xm[1] * w_in[k * 3 + 1] + xm[2] * w_in[k * 3 + 2];
      myH[s][idx] = (_Float16)(v > 0.f ? v : 0.f);
    }
  }
  v16h A0[SLABS], A1[SLABS];
#pragma unroll
  for (int s = 0; s < SLABS; ++s) loadA(myH[s], lane, A0[s], A1[s]);

  v8f po[SLABS][4];
#pragma unroll
  for (int s = 0; s < SLABS; ++s)
#pragma unroll
    for (int nt = 0; nt < 4; ++nt)
#pragma unroll
      for (int r = 0; r < 8; ++r) po[s][nt][r] = 0.f;

  const int nlayers = (pass < NLYR) ? (pass + 1) : NLYR;
  for (int i = 0; i < nlayers; ++i) {
    const bool poolLayer = (pass < NLYR) && (i == pass);

    __syncthreads();   // WAR on Bst from previous layer's reads (also covers prologue)
    stageB(&Bst[0][0][0][0], lw + (size_t)i * HID * HID, HID, tid);
    if (!poolLayer) {
      stageB(&Bst[1][0][0][0], gw + (size_t)i * HID * 2 * HID, 2 * HID, tid);
      if (pass == NLYR) stageB(&Bst[2][0][0][0], wo + (size_t)i * HID, NLYR * HID, tid);
    }
    __syncthreads();

    // ---- h1 = relu(h @ lyr_w[i]^T + lyr_b[i]) ----
    v8f C[SLABS][4];
#pragma unroll
    for (int nt = 0; nt < 4; ++nt) {
      float bias = lb[i * HID + nt * 16 + (lane & 15)];
      v16h B0 = loadBst(&Bst[0][0][0][0], nt * 2, lane);
      v16h B1 = loadBst(&Bst[0][0][0][0], nt * 2 + 1, lane);
#pragma unroll
      for (int s = 0; s < SLABS; ++s) {
        v8f c;
#pragma unroll
        for (int r = 0; r < 8; ++r) c[r] = bias;
        c = wmma(A0[s], B0, c);
        c = wmma(A1[s], B1, c);
        C[s][nt] = c;
      }
    }

    if (poolLayer) {
      // pool max_l relu(h1_pass): relu folded via mx init 0 (values >= 0)
#pragma unroll
      for (int nt = 0; nt < 4; ++nt) {
        float mx = 0.f;
#pragma unroll
        for (int s = 0; s < SLABS; ++s)
#pragma unroll
          for (int r = 0; r < 8; ++r) { float f = C[s][nt][r]; mx = f > mx ? f : mx; }
        float o = __shfl_xor(mx, 16, 32);
        mx = o > mx ? o : mx;
        if (lane < 16) atomicMax(&red[nt * 16 + lane], __float_as_uint(mx));
      }
      break;
    }

    // relayout h1 -> A frags (same-wave LDS roundtrip; DS ops are in-order per wave)
#pragma unroll
    for (int s = 0; s < SLABS; ++s) {
#pragma unroll
      for (int nt = 0; nt < 4; ++nt) storeC(myH[s], lane, nt * 16, C[s][nt]);
      loadA(myH[s], lane, A0[s], A1[s]);
    }

    // ---- h2 = relu(h1 @ Wg[:, :64]^T + gbias[i]) ----
#pragma unroll
    for (int nt = 0; nt < 4; ++nt) {
      float bias = gbias[i][nt * 16 + (lane & 15)];
      v16h B0 = loadBst(&Bst[1][0][0][0], nt * 2, lane);
      v16h B1 = loadBst(&Bst[1][0][0][0], nt * 2 + 1, lane);
#pragma unroll
      for (int s = 0; s < SLABS; ++s) {
        v8f c;
#pragma unroll
        for (int r = 0; r < 8; ++r) c[r] = bias;
        c = wmma(A0[s], B0, c);
        c = wmma(A1[s], B1, c);
        C[s][nt] = c;
      }
    }
#pragma unroll
    for (int s = 0; s < SLABS; ++s) {
#pragma unroll
      for (int nt = 0; nt < 4; ++nt) storeC(myH[s], lane, nt * 16, C[s][nt]);
      loadA(myH[s], lane, A0[s], A1[s]);   // h2_i: next layer input AND proj_out operand
    }

    if (pass == NLYR) {
      // po += h2_i @ Wout[:, 64i:64i+64]^T
#pragma unroll
      for (int nt = 0; nt < 4; ++nt) {
        v16h B0 = loadBst(&Bst[2][0][0][0], nt * 2, lane);
        v16h B1 = loadBst(&Bst[2][0][0][0], nt * 2 + 1, lane);
#pragma unroll
        for (int s = 0; s < SLABS; ++s) {
          po[s][nt] = wmma(A0[s], B0, po[s][nt]);
          po[s][nt] = wmma(A1[s], B1, po[s][nt]);
        }
      }
    }
  }

  if (pass < NLYR) {
    __syncthreads();
    if (tid < HID)
      atomicMax(g_ws + ((size_t)pass * NBATCH + b) * HID + tid, red[tid]);
  } else {
    // signed-order uint keys: f>=0 -> bits|0x80000000 ; f<0 -> ~bits  (all keys > 0)
#pragma unroll
    for (int nt = 0; nt < 4; ++nt) {
      float mx = -3.4e38f;
#pragma unroll
      for (int s = 0; s < SLABS; ++s)
#pragma unroll
        for (int r = 0; r < 8; ++r) { float f = po[s][nt][r]; mx = f > mx ? f : mx; }
      float o = __shfl_xor(mx, 16, 32);
      mx = o > mx ? o : mx;
      unsigned bb = __float_as_uint(mx);
      unsigned key = (bb & 0x80000000u) ? ~bb : (bb | 0x80000000u);
      if (lane < 16) atomicMax(&red[nt * 16 + lane], key);
    }
    __syncthreads();
    if (tid < HID)
      atomicMax(o_ws + (size_t)b * HID + tid, red[tid]);
  }
}

__global__ void pe_init(unsigned* __restrict__ ws, int n) {
  int i = blockIdx.x * blockDim.x + threadIdx.x;
  if (i < n) ws[i] = 0u;
}

__global__ void pe_decode(const unsigned* __restrict__ o_ws, const float* __restrict__ b_out,
                          float* __restrict__ out) {
  int i = blockIdx.x * blockDim.x + threadIdx.x;
  if (i < NBATCH * HID) {
    unsigned key = o_ws[i];
    unsigned bb = (key & 0x80000000u) ? (key ^ 0x80000000u) : ~key;
    out[i] = __uint_as_float(bb) + b_out[i & (HID - 1)];
  }
}

extern "C" void kernel_launch(void* const* d_in, const int* in_sizes, int n_in,
                              void* d_out, int out_size, void* d_ws, size_t ws_size,
                              hipStream_t stream) {
  (void)in_sizes; (void)n_in; (void)out_size; (void)ws_size;
  const float* x   = (const float*)d_in[0];
  const float* wIn = (const float*)d_in[1];
  const float* bIn = (const float*)d_in[2];
  const float* lwp = (const float*)d_in[3];
  const float* lbp = (const float*)d_in[4];
  const float* gwp = (const float*)d_in[5];
  const float* gbp = (const float*)d_in[6];
  const float* wop = (const float*)d_in[7];
  const float* bop = (const float*)d_in[8];

  unsigned* g_ws = (unsigned*)d_ws;                      // [4][256][64] relu-max bits
  unsigned* o_ws = g_ws + (size_t)NLYR * NBATCH * HID;   // [256][64] signed keys

  const int tot = (NLYR + 1) * NBATCH * HID;
  pe_init<<<(tot + 255) / 256, 256, 0, stream>>>(g_ws, tot);

  dim3 grid(CHUNKS, NBATCH);
  for (int pass = 0; pass <= NLYR; ++pass)
    pe_pass<<<grid, THREADS, 0, stream>>>(x, wIn, bIn, lwp, lbp, gwp, gbp, wop,
                                          g_ws, o_ws, pass);

  pe_decode<<<(NBATCH * HID + 255) / 256, 256, 0, stream>>>(o_ws, bop, (float*)d_out);
}
